// YourNLPModel_55911884259863
// MI455X (gfx1250) — matrix operand order repaired and checked by reading, not compiled, not run on hardware
//
#include <hip/hip_runtime.h>

typedef __attribute__((ext_vector_type(16))) _Float16 v16h;
typedef __attribute__((ext_vector_type(8)))  float    v8f;

#define S_LEN   2048
#define BATCH   64
#define DMODEL  128
#define HID     64
#define G4      256          // 4*H gate width
#define APK_ST  68           // padded uint stride for packed A rows (64 kp + 4 pad)
#define HPK_ST  36           // padded uint stride for packed h rows (32 kp + 4 pad)
#define NB_WG   16           // batches per scan workgroup (4 independent workgroups)

union Frag {
  v16h h;
  unsigned int w[8];
  uint4 q[2];
};

__device__ __forceinline__ unsigned int pkf16(float a, float b) {
  union { _Float16 h[2]; unsigned int u; } t;
  t.h[0] = (_Float16)a; t.h[1] = (_Float16)b; return t.u;
}
__device__ __forceinline__ unsigned short f16bits(float a) {
  union { _Float16 h; unsigned short u; } t; t.h = (_Float16)a; return t.u;
}
__device__ __forceinline__ float f16val(unsigned short u) {
  union { unsigned short us; _Float16 h; } t; t.us = u; return (float)t.h;
}

#if __has_builtin(__builtin_amdgcn_tanh_f32)
  #define TANHF(x) __builtin_amdgcn_tanh_f32(x)
#elif __has_builtin(__builtin_amdgcn_tanhf)
  #define TANHF(x) __builtin_amdgcn_tanhf(x)
#else
  #define TANHF(x) (1.0f - 2.0f / (1.0f + __expf(2.0f * (x))))
#endif
__device__ __forceinline__ float sigmoidf(float x) { return 0.5f * TANHF(0.5f * x) + 0.5f; }

// ---------------------------------------------------------------------------
// Kernel 1: embedding gather + [64 x 128] x [128 x 256] WMMA tile GEMM per block.
// Output xg stored time-major f16: xg[((s*64)+b)*256 + g], bias folded in.
// ---------------------------------------------------------------------------
__global__ __launch_bounds__(256) void xg_gemm_kernel(
    const int* __restrict__ x, const float* __restrict__ emb,
    const float* __restrict__ W_ih, const float* __restrict__ b_ih,
    const float* __restrict__ b_hh, unsigned short* __restrict__ xg)
{
  __shared__ unsigned int Bpk[32 * G4];      // 32 KB: packed-f16 W_ih^T, half of K
  __shared__ unsigned int Apk[64 * APK_ST];  // 17.4 KB: packed-f16 gathered emb rows
  __shared__ float bias_sh[G4];
  __shared__ int idx_sh[64];

  const int tid = threadIdx.x;
  const int wgm = blockIdx.x * 64;           // first global row m = b*S + s

  if (tid < 64) idx_sh[tid] = x[wgm + tid];
  if (tid < G4) bias_sh[tid] = b_ih[tid] + b_hh[tid];
  __syncthreads();

  // gather + f16-pack A rows (all of K)
  for (int i = tid; i < 64 * 64; i += 256) {
    const int r = i >> 6, kp = i & 63;
    const float2 e = *(const float2*)(emb + (size_t)idx_sh[r] * DMODEL + kp * 2);
    Apk[r * APK_ST + kp] = pkf16(e.x, e.y);
  }

  const int lane = tid & 31;
  const int wv   = tid >> 5;   // 8 waves
  const int mt   = wv & 3;     // M tile 0..3
  const int nh   = wv >> 2;    // N half: nt = nh*8 + j
  const int n    = lane & 15;
  const int hi   = lane >> 4;

  v8f acc[8] = {};

  for (int kh = 0; kh < 2; ++kh) {           // K in two 64-wide phases (LDS budget)
    __syncthreads();                         // prior phase done with Bpk; Apk ready
    for (int i = tid; i < 32 * G4; i += 256) {
      const int kpl = i >> 8, g = i & 255;
      const float2 wv2 = *(const float2*)(W_ih + g * DMODEL + (kh * 32 + kpl) * 2);
      Bpk[kpl * G4 + g] = pkf16(wv2.x, wv2.y);
    }
    __syncthreads();

    for (int ks2 = 0; ks2 < 2; ++ks2) {
      const int ks = kh * 2 + ks2;
      Frag a;                                 // A frag: lane = M, hi = K half
      const int m = mt * 16 + n;
      a.q[0] = *(const uint4*)&Apk[m * APK_ST + ks * 16 + hi * 8];
      a.q[1] = *(const uint4*)&Apk[m * APK_ST + ks * 16 + hi * 8 + 4];
      #pragma unroll
      for (int j = 0; j < 8; ++j) {
        const int g = (nh * 8 + j) * 16 + n;  // B frag: lane = N, hi = K group
        Frag b;
        #pragma unroll
        for (int p = 0; p < 8; ++p)
          b.w[p] = Bpk[(ks2 * 16 + hi * 8 + p) * G4 + g];
        acc[j] = __builtin_amdgcn_wmma_f32_16x16x32_f16(
            false, a.h, false, b.h, (short)0, acc[j], false, false);
      }
    }
  }

  // bias + store time-major f16
  #pragma unroll
  for (int j = 0; j < 8; ++j) {
    const int g = (nh * 8 + j) * 16 + n;
    const float bv = bias_sh[g];
    #pragma unroll
    for (int r = 0; r < 8; ++r) {
      const int mrow = wgm + mt * 16 + r + 8 * hi;  // C/D: VGPR r = row, hi = row half
      const int b = mrow >> 11;                     // m / 2048
      const int s = mrow & (S_LEN - 1);
      xg[((size_t)s * BATCH + b) * G4 + g] = f16bits(acc[j][r] + bv);
    }
  }
}

// ---------------------------------------------------------------------------
// Kernel 2: sequential LSTM scan, batch-parallel across 4 independent
// workgroups (16 batches / 1 M-tile each, 4 waves = 128 threads, 1 wave/SIMD).
// W_hh B-fragments resident in VGPRs for all 2048 steps; h double-buffered as
// packed f16 in LDS; c in VGPRs. xg is software-pipelined: step t issues the
// register loads for step t+1 right after the barrier (full step of latency
// hiding), plus a global_prefetch for t+2.
// Wave q handles gate tiles nt = jt*4+q so all four gates of each (b,hidx)
// cell are lane-local for the pointwise LSTM update.
// ---------------------------------------------------------------------------
__global__ __launch_bounds__(128) void lstm_scan_kernel(
    const float* __restrict__ W_hh, const unsigned short* __restrict__ xg,
    float* __restrict__ hlast)
{
  __shared__ unsigned int hpk[2][NB_WG * HPK_ST];  // 2 x 2.3 KB packed-f16 h

  const int tid  = threadIdx.x;
  const int lane = tid & 31;
  const int q    = tid >> 5;      // 4 waves: hidden tile q
  const int n    = lane & 15;
  const int hi   = lane >> 4;
  const int wg   = blockIdx.x;    // batch slice [wg*16, wg*16+16)

  // Preload W_hh^T B-fragments (f16) — resident in VGPRs for the whole scan.
  unsigned int bf[4][2][8];
  #pragma unroll
  for (int jt = 0; jt < 4; ++jt) {
    const int g = (jt * 4 + q) * 16 + n;
    #pragma unroll
    for (int ks = 0; ks < 2; ++ks) {
      #pragma unroll
      for (int p = 0; p < 8; ++p) {
        const int k0 = ks * 32 + hi * 16 + 2 * p;
        const float2 wv2 = *(const float2*)(W_hh + g * HID + k0);
        bf[jt][ks][p] = pkf16(wv2.x, wv2.y);
      }
    }
  }

  for (int i = tid; i < NB_WG * HPK_ST; i += 128) hpk[0][i] = 0;  // h0 = 0
  float c[8];
  #pragma unroll
  for (int r = 0; r < 8; ++r) c[r] = 0.0f;

  const size_t slice = (size_t)wg * NB_WG * G4;   // this WG's batch slice offset

  // register loader for one step's xg values (this wave's 4 gate tiles)
  auto load_xg = [&](int t, float (&dst)[4][8]) {
    const size_t base = (size_t)t * BATCH * G4 + slice;
    #pragma unroll
    for (int jt = 0; jt < 4; ++jt) {
      const int g = jt * 64 + q * 16 + n;
      #pragma unroll
      for (int r = 0; r < 8; ++r) {
        const int bl = r + 8 * hi;
        dst[jt][r] = f16val(xg[base + (size_t)bl * G4 + g]);
      }
    }
  };

  auto step = [&](int t, float (&cur)[4][8], float (&nxt)[4][8]) {
    __syncthreads();                       // hpk[t&1] writes + cur loads complete
    if (t + 1 < S_LEN) load_xg(t + 1, nxt);   // pipeline next step into regs
    if (t + 2 < S_LEN)                        // warm WGP$ for t+2 (8 KB slice)
      __builtin_prefetch((const void*)(xg + (size_t)(t + 2) * BATCH * G4 + slice + tid * 32), 0, 1);

    const int rd = t & 1, wr = rd ^ 1;

    v8f acc[4];
    #pragma unroll
    for (int jt = 0; jt < 4; ++jt)
      #pragma unroll
      for (int r = 0; r < 8; ++r) acc[jt][r] = cur[jt][r];

    // recurrent GEMM: h (f16, LDS) x W_hh^T (f16, VGPR) -> acc
    #pragma unroll
    for (int ks = 0; ks < 2; ++ks) {
      Frag a;                              // A frag: lane = local batch row, hi = K half
      a.q[0] = *(const uint4*)&hpk[rd][n * HPK_ST + ks * 16 + hi * 8];
      a.q[1] = *(const uint4*)&hpk[rd][n * HPK_ST + ks * 16 + hi * 8 + 4];
      #pragma unroll
      for (int jt = 0; jt < 4; ++jt) {
        Frag b;
        #pragma unroll
        for (int p = 0; p < 8; ++p) b.w[p] = bf[jt][ks][p];
        acc[jt] = __builtin_amdgcn_wmma_f32_16x16x32_f16(
            false, a.h, false, b.h, (short)0, acc[jt], false, false);
      }
    }

    // gates + state update; each (lane,r) owns one (bl,hidx) cell
    const int hidx = q * 16 + n;
    #pragma unroll
    for (int r = 0; r < 8; ++r) {
      const float i_ = sigmoidf(acc[0][r]);
      const float f_ = sigmoidf(acc[1][r]);
      const float g_ = TANHF(acc[2][r]);
      const float o_ = sigmoidf(acc[3][r]);
      c[r] = f_ * c[r] + i_ * g_;
      const float h = o_ * TANHF(c[r]);
      const int bl = r + 8 * hi;
      ((unsigned short*)&hpk[wr][bl * HPK_ST])[hidx] = f16bits(h);
      if (t == S_LEN - 1) hlast[(wg * NB_WG + bl) * HID + hidx] = h;
    }
  };

  float xa[4][8], xb[4][8];
  load_xg(0, xa);
  for (int t = 0; t < S_LEN; t += 2) {   // unroll-2: register double-buffer swap
    step(t,     xa, xb);
    step(t + 1, xb, xa);
  }
}

// ---------------------------------------------------------------------------
// Kernel 3: tiny FC head  out[b][o] = b_fc[o] + h_last[b,:] . W_fc[o,:]
// ---------------------------------------------------------------------------
__global__ __launch_bounds__(256) void fc_kernel(
    const float* __restrict__ hlast, const float* __restrict__ W_fc,
    const float* __restrict__ b_fc, float* __restrict__ out)
{
  const int i = blockIdx.x * 256 + threadIdx.x;
  if (i >= BATCH * 10) return;
  const int b = i / 10, o = i % 10;
  float s = b_fc[o];
  #pragma unroll
  for (int k = 0; k < HID; ++k) s += hlast[b * HID + k] * W_fc[o * HID + k];
  out[i] = s;
}

extern "C" void kernel_launch(void* const* d_in, const int* in_sizes, int n_in,
                              void* d_out, int out_size, void* d_ws, size_t ws_size,
                              hipStream_t stream) {
  const int*   x    = (const int*)d_in[0];
  const float* emb  = (const float*)d_in[1];
  const float* W_ih = (const float*)d_in[2];
  const float* W_hh = (const float*)d_in[3];
  const float* b_ih = (const float*)d_in[4];
  const float* b_hh = (const float*)d_in[5];
  const float* W_fc = (const float*)d_in[6];
  const float* b_fc = (const float*)d_in[7];
  float* out = (float*)d_out;

  const size_t xg_bytes = (size_t)S_LEN * BATCH * G4 * sizeof(unsigned short);
  const size_t need = xg_bytes + (size_t)BATCH * HID * sizeof(float);
  if (ws_size < need) return;  // deterministic no-op if scratch undersized

  unsigned short* xg = (unsigned short*)d_ws;
  float* hlast = (float*)((char*)d_ws + xg_bytes);

  xg_gemm_kernel<<<(BATCH * S_LEN) / 64, 256, 0, stream>>>(x, emb, W_ih, b_ih, b_hh, xg);
  lstm_scan_kernel<<<BATCH / NB_WG, 128, 0, stream>>>(W_hh, xg, hlast);
  fc_kernel<<<(BATCH * 10 + 255) / 256, 256, 0, stream>>>(hlast, W_fc, b_fc, out);
}
